// Embedder_83846351552792
// MI455X (gfx1250) — compile-verified, matrix operand order checked by reading, not run on hardware
//
#include <hip/hip_runtime.h>
#include <math.h>

typedef __attribute__((ext_vector_type(16))) __bf16 v16bf;
typedef __attribute__((ext_vector_type(8)))  __bf16 v8bf;
typedef __attribute__((ext_vector_type(4)))  __bf16 v4bf;
typedef __attribute__((ext_vector_type(8)))  float  v8f;
typedef __attribute__((ext_vector_type(4)))  float  f32x4;

static constexpr int LL = 8, NN = 1024, DD = 2048, KK = 4096;
static constexpr int BM = 128;   // x rows per block
static constexpr int BN = 128;   // centroid rows per block
static constexpr int KC = 32;    // D chunk (bf16 WMMA K)
static constexpr int LDT = 40;   // padded LDS row stride in bf16 (80B, multiple of 16B)

// ---------------- fused convert + row-norm: one wave32 per row ----------------
// hi/lo split of scale*src (scale=-2 for x folds the -2 exactly); norm of src.
__global__ __launch_bounds__(256) void convert_norm_kernel(const float* __restrict__ src,
                                                           __bf16* __restrict__ hi,
                                                           __bf16* __restrict__ lo,
                                                           float* __restrict__ norm,
                                                           int rows, float scale) {
    int row  = blockIdx.x * 8 + (threadIdx.x >> 5);
    int lane = threadIdx.x & 31;
    if (row >= rows) return;
    const float* p  = src + (size_t)row * DD;
    __bf16*      ph = hi  + (size_t)row * DD;
    __bf16*      pl = lo  + (size_t)row * DD;
    float s = 0.0f;
    #pragma unroll 4
    for (int d = 0; d < DD; d += 128) {
        int idx = d + lane * 4;
        f32x4 v = *(const f32x4*)(p + idx);
        s += v.x * v.x + v.y * v.y + v.z * v.z + v.w * v.w;
        float f0 = scale * v.x, f1 = scale * v.y, f2 = scale * v.z, f3 = scale * v.w;
        __bf16 h0 = (__bf16)f0, h1 = (__bf16)f1, h2 = (__bf16)f2, h3 = (__bf16)f3;
        *(v4bf*)(ph + idx) = (v4bf){h0, h1, h2, h3};
        *(v4bf*)(pl + idx) = (v4bf){(__bf16)(f0 - (float)h0), (__bf16)(f1 - (float)h1),
                                    (__bf16)(f2 - (float)h2), (__bf16)(f3 - (float)h3)};
    }
    #pragma unroll
    for (int off = 16; off >= 1; off >>= 1)
        s += __shfl_down(s, off, 32);
    if (lane == 0) norm[row] = s;
}

// ---------------- plain row-norm (fallback path) ----------------
__global__ __launch_bounds__(256) void row_norm_kernel(const float* __restrict__ src,
                                                       float* __restrict__ dst,
                                                       int rows) {
    int row  = blockIdx.x * 8 + (threadIdx.x >> 5);
    int lane = threadIdx.x & 31;
    if (row >= rows) return;
    const float* p = src + (size_t)row * DD;
    float s = 0.0f;
    #pragma unroll
    for (int d = 0; d < DD; d += 128) {
        f32x4 v = *(const f32x4*)(p + d + lane * 4);
        s += v.x * v.x + v.y * v.y + v.z * v.z + v.w * v.w;
    }
    #pragma unroll
    for (int off = 16; off >= 1; off >>= 1)
        s += __shfl_down(s, off, 32);
    if (lane == 0) dst[row] = s;
}

// ============================================================================
// Optimized path: pre-converted bf16 hi/lo planes -> zero conversion VALU in
// the GEMM loop; global loads prefetched one chunk ahead of the WMMAs.
// ============================================================================
__global__ __launch_bounds__(256) void dist_wmma_pre_kernel(const __bf16* __restrict__ xhi,
                                                            const __bf16* __restrict__ xlo,
                                                            const __bf16* __restrict__ chi,
                                                            const __bf16* __restrict__ clo,
                                                            const float* __restrict__ x2,
                                                            const float* __restrict__ c2,
                                                            float* __restrict__ out) {
    __shared__ __bf16 sAhi[BM * LDT];
    __shared__ __bf16 sAlo[BM * LDT];
    __shared__ __bf16 sBhi[BN * LDT];
    __shared__ __bf16 sBlo[BN * LDT];

    const int l  = blockIdx.z;
    const int nb = blockIdx.y * BM;
    const int kb = blockIdx.x * BN;

    const int tid  = threadIdx.x;
    const int wave = tid >> 5;
    const int lane = tid & 31;
    const int grp  = lane >> 4;
    const int l16  = lane & 15;

    const int wm = (wave & 1) * 64;
    const int wn = (wave >> 1) * 32;

    const __bf16* xh = xhi + (size_t)l * NN * DD + (size_t)nb * DD;
    const __bf16* xw = xlo + (size_t)l * NN * DD + (size_t)nb * DD;
    const __bf16* ch = chi + (size_t)l * KK * DD + (size_t)kb * DD;
    const __bf16* cw = clo + (size_t)l * KK * DD + (size_t)kb * DD;

    v8f acc[4][2];
    #pragma unroll
    for (int mi = 0; mi < 4; ++mi)
        #pragma unroll
        for (int ni = 0; ni < 2; ++ni)
            acc[mi][ni] = (v8f){0.f, 0.f, 0.f, 0.f, 0.f, 0.f, 0.f, 0.f};

    // cooperative loader: 4 threads cover one row's 32 bf16 (64B); 2 row-groups
    const int lr = tid >> 2;          // 0..63
    const int lc = (tid & 3) * 8;     // 0,8,16,24

    // fragment K-striping (ISA 7.12.2, 16-bit A/B 16x32)
    const int c0 = grp ? 8 : 0;
    const int c1 = grp ? 24 : 16;

    v8bf rAh[2], rAl[2], rBh[2], rBl[2];
    auto gload = [&](int d0) {
        #pragma unroll
        for (int g = 0; g < 2; ++g) {
            size_t off = (size_t)(lr + 64 * g) * DD + d0 + lc;
            rAh[g] = *(const v8bf*)(xh + off);
            rAl[g] = *(const v8bf*)(xw + off);
            rBh[g] = *(const v8bf*)(ch + off);
            rBl[g] = *(const v8bf*)(cw + off);
        }
    };
    gload(0);

    for (int d0 = 0; d0 < DD; d0 += KC) {
        __syncthreads();   // previous chunk's fragment reads complete
        #pragma unroll
        for (int g = 0; g < 2; ++g) {
            int r = lr + 64 * g;
            *(v8bf*)(&sAhi[r * LDT + lc]) = rAh[g];
            *(v8bf*)(&sAlo[r * LDT + lc]) = rAl[g];
            *(v8bf*)(&sBhi[r * LDT + lc]) = rBh[g];
            *(v8bf*)(&sBlo[r * LDT + lc]) = rBl[g];
        }
        __syncthreads();

        if (d0 + KC < DD) gload(d0 + KC);   // prefetch next chunk behind the WMMAs

        v16bf ah[4], al[4], bh[2], bl[2];
        #pragma unroll
        for (int mi = 0; mi < 4; ++mi) {
            const __bf16* rp = &sAhi[(wm + mi * 16 + l16) * LDT];
            v8bf p0 = *(const v8bf*)(rp + c0);
            v8bf p1 = *(const v8bf*)(rp + c1);
            ah[mi] = __builtin_shufflevector(p0, p1, 0,1,2,3,4,5,6,7,8,9,10,11,12,13,14,15);
            rp = &sAlo[(wm + mi * 16 + l16) * LDT];
            p0 = *(const v8bf*)(rp + c0);
            p1 = *(const v8bf*)(rp + c1);
            al[mi] = __builtin_shufflevector(p0, p1, 0,1,2,3,4,5,6,7,8,9,10,11,12,13,14,15);
        }
        #pragma unroll
        for (int ni = 0; ni < 2; ++ni) {
            const __bf16* rp = &sBhi[(wn + ni * 16 + l16) * LDT];
            v8bf p0 = *(const v8bf*)(rp + c0);
            v8bf p1 = *(const v8bf*)(rp + c1);
            bh[ni] = __builtin_shufflevector(p0, p1, 0,1,2,3,4,5,6,7,8,9,10,11,12,13,14,15);
            rp = &sBlo[(wn + ni * 16 + l16) * LDT];
            p0 = *(const v8bf*)(rp + c0);
            p1 = *(const v8bf*)(rp + c1);
            bl[ni] = __builtin_shufflevector(p0, p1, 0,1,2,3,4,5,6,7,8,9,10,11,12,13,14,15);
        }

        #pragma unroll
        for (int mi = 0; mi < 4; ++mi) {
            #pragma unroll
            for (int ni = 0; ni < 2; ++ni) {
                acc[mi][ni] = __builtin_amdgcn_wmma_f32_16x16x32_bf16(
                    false, ah[mi], false, bh[ni], (short)0, acc[mi][ni], false, false);
                acc[mi][ni] = __builtin_amdgcn_wmma_f32_16x16x32_bf16(
                    false, ah[mi], false, bl[ni], (short)0, acc[mi][ni], false, false);
                acc[mi][ni] = __builtin_amdgcn_wmma_f32_16x16x32_bf16(
                    false, al[mi], false, bh[ni], (short)0, acc[mi][ni], false, false);
            }
        }
    }

    float* ol = out + (size_t)l * NN * KK;
    const float* x2l = x2 + l * NN + nb;
    const float* c2l = c2 + l * KK + kb;
    #pragma unroll
    for (int mi = 0; mi < 4; ++mi) {
        #pragma unroll
        for (int ni = 0; ni < 2; ++ni) {
            const int colLocal = wn + ni * 16 + l16;
            const float cc = c2l[colLocal];
            #pragma unroll
            for (int j = 0; j < 8; ++j) {
                const int rowLocal = wm + mi * 16 + grp * 8 + j;
                float d2 = x2l[rowLocal] + cc + acc[mi][ni][j];  // acc holds -2*x.c
                ol[(size_t)(nb + rowLocal) * KK + (kb + colLocal)] =
                    sqrtf(fmaxf(d2, 1e-12f));
            }
        }
    }
}

// ============================================================================
// Fallback path: on-the-fly f32 -> bf16 hi/lo conversion (round-1 kernel),
// used only if the workspace is too small for the pre-converted planes.
// ============================================================================
__global__ __launch_bounds__(256) void dist_wmma_kernel(const float* __restrict__ x,
                                                        const float* __restrict__ cen,
                                                        const float* __restrict__ x2,
                                                        const float* __restrict__ c2,
                                                        float* __restrict__ out) {
    __shared__ __bf16 sAhi[BM * LDT];
    __shared__ __bf16 sAlo[BM * LDT];
    __shared__ __bf16 sBhi[BN * LDT];
    __shared__ __bf16 sBlo[BN * LDT];

    const int l  = blockIdx.z;
    const int nb = blockIdx.y * BM;
    const int kb = blockIdx.x * BN;

    const int tid  = threadIdx.x;
    const int wave = tid >> 5;
    const int lane = tid & 31;
    const int grp  = lane >> 4;
    const int l16  = lane & 15;

    const int wm = (wave & 1) * 64;
    const int wn = (wave >> 1) * 32;

    const float* xl = x   + (size_t)l * NN * DD + (size_t)nb * DD;
    const float* cl = cen + (size_t)l * KK * DD + (size_t)kb * DD;

    v8f acc[4][2];
    #pragma unroll
    for (int mi = 0; mi < 4; ++mi)
        #pragma unroll
        for (int ni = 0; ni < 2; ++ni)
            acc[mi][ni] = (v8f){0.f, 0.f, 0.f, 0.f, 0.f, 0.f, 0.f, 0.f};

    const int tr = tid >> 3;
    const int tc = (tid & 7) * 4;
    const int c0 = grp ? 8 : 0;
    const int c1 = grp ? 24 : 16;

    for (int d0 = 0; d0 < DD; d0 += KC) {
        #pragma unroll
        for (int rr = 0; rr < BM; rr += 32) {
            int r = tr + rr;
            f32x4 v = *(const f32x4*)(xl + (size_t)r * DD + d0 + tc);
            float f0 = -2.0f * v.x, f1 = -2.0f * v.y, f2 = -2.0f * v.z, f3 = -2.0f * v.w;
            __bf16 h0 = (__bf16)f0, h1 = (__bf16)f1, h2 = (__bf16)f2, h3 = (__bf16)f3;
            *(v4bf*)(&sAhi[r * LDT + tc]) = (v4bf){h0, h1, h2, h3};
            *(v4bf*)(&sAlo[r * LDT + tc]) =
                (v4bf){(__bf16)(f0 - (float)h0), (__bf16)(f1 - (float)h1),
                       (__bf16)(f2 - (float)h2), (__bf16)(f3 - (float)h3)};
        }
        #pragma unroll
        for (int rr = 0; rr < BN; rr += 32) {
            int r = tr + rr;
            f32x4 v = *(const f32x4*)(cl + (size_t)r * DD + d0 + tc);
            __bf16 h0 = (__bf16)v.x, h1 = (__bf16)v.y, h2 = (__bf16)v.z, h3 = (__bf16)v.w;
            *(v4bf*)(&sBhi[r * LDT + tc]) = (v4bf){h0, h1, h2, h3};
            *(v4bf*)(&sBlo[r * LDT + tc]) =
                (v4bf){(__bf16)(v.x - (float)h0), (__bf16)(v.y - (float)h1),
                       (__bf16)(v.z - (float)h2), (__bf16)(v.w - (float)h3)};
        }
        __syncthreads();

        v16bf ah[4], al[4], bh[2], bl[2];
        #pragma unroll
        for (int mi = 0; mi < 4; ++mi) {
            const __bf16* rp = &sAhi[(wm + mi * 16 + l16) * LDT];
            v8bf p0 = *(const v8bf*)(rp + c0);
            v8bf p1 = *(const v8bf*)(rp + c1);
            ah[mi] = __builtin_shufflevector(p0, p1, 0,1,2,3,4,5,6,7,8,9,10,11,12,13,14,15);
            rp = &sAlo[(wm + mi * 16 + l16) * LDT];
            p0 = *(const v8bf*)(rp + c0);
            p1 = *(const v8bf*)(rp + c1);
            al[mi] = __builtin_shufflevector(p0, p1, 0,1,2,3,4,5,6,7,8,9,10,11,12,13,14,15);
        }
        #pragma unroll
        for (int ni = 0; ni < 2; ++ni) {
            const __bf16* rp = &sBhi[(wn + ni * 16 + l16) * LDT];
            v8bf p0 = *(const v8bf*)(rp + c0);
            v8bf p1 = *(const v8bf*)(rp + c1);
            bh[ni] = __builtin_shufflevector(p0, p1, 0,1,2,3,4,5,6,7,8,9,10,11,12,13,14,15);
            rp = &sBlo[(wn + ni * 16 + l16) * LDT];
            p0 = *(const v8bf*)(rp + c0);
            p1 = *(const v8bf*)(rp + c1);
            bl[ni] = __builtin_shufflevector(p0, p1, 0,1,2,3,4,5,6,7,8,9,10,11,12,13,14,15);
        }

        #pragma unroll
        for (int mi = 0; mi < 4; ++mi) {
            #pragma unroll
            for (int ni = 0; ni < 2; ++ni) {
                acc[mi][ni] = __builtin_amdgcn_wmma_f32_16x16x32_bf16(
                    false, ah[mi], false, bh[ni], (short)0, acc[mi][ni], false, false);
                acc[mi][ni] = __builtin_amdgcn_wmma_f32_16x16x32_bf16(
                    false, ah[mi], false, bl[ni], (short)0, acc[mi][ni], false, false);
                acc[mi][ni] = __builtin_amdgcn_wmma_f32_16x16x32_bf16(
                    false, al[mi], false, bh[ni], (short)0, acc[mi][ni], false, false);
            }
        }
        __syncthreads();
    }

    float* ol = out + (size_t)l * NN * KK;
    const float* x2l = x2 + l * NN + nb;
    const float* c2l = c2 + l * KK + kb;
    #pragma unroll
    for (int mi = 0; mi < 4; ++mi) {
        #pragma unroll
        for (int ni = 0; ni < 2; ++ni) {
            const int colLocal = wn + ni * 16 + l16;
            const float cc = c2l[colLocal];
            #pragma unroll
            for (int j = 0; j < 8; ++j) {
                const int rowLocal = wm + mi * 16 + grp * 8 + j;
                float d2 = x2l[rowLocal] + cc + acc[mi][ni][j];
                ol[(size_t)(nb + rowLocal) * KK + (kb + colLocal)] =
                    sqrtf(fmaxf(d2, 1e-12f));
            }
        }
    }
}

extern "C" void kernel_launch(void* const* d_in, const int* in_sizes, int n_in,
                              void* d_out, int out_size, void* d_ws, size_t ws_size,
                              hipStream_t stream) {
    (void)in_sizes; (void)n_in; (void)out_size;
    const float* x   = (const float*)d_in[0];
    const float* cen = (const float*)d_in[1];
    float* out = (float*)d_out;

    const size_t xElems = (size_t)LL * NN * DD;   // 16.78M
    const size_t cElems = (size_t)LL * KK * DD;   // 67.1M
    const size_t normFloats = (size_t)LL * NN + (size_t)LL * KK;
    const size_t need = normFloats * sizeof(float) + 2 * (xElems + cElems) * sizeof(__bf16);

    float* x2 = (float*)d_ws;
    float* c2 = x2 + LL * NN;
    dim3 grid(KK / BN, NN / BM, LL);   // (32, 8, 8)

    if (ws_size >= need) {
        __bf16* xhi = (__bf16*)(c2 + LL * KK);
        __bf16* xlo = xhi + xElems;
        __bf16* chi = xlo + xElems;
        __bf16* clo = chi + cElems;
        convert_norm_kernel<<<(LL * NN) / 8, 256, 0, stream>>>(x,   xhi, xlo, x2, LL * NN, -2.0f);
        convert_norm_kernel<<<(LL * KK) / 8, 256, 0, stream>>>(cen, chi, clo, c2, LL * KK, 1.0f);
        dist_wmma_pre_kernel<<<grid, 256, 0, stream>>>(xhi, xlo, chi, clo, x2, c2, out);
    } else {
        row_norm_kernel<<<(LL * NN) / 8, 256, 0, stream>>>(x, x2, LL * NN);
        row_norm_kernel<<<(LL * KK) / 8, 256, 0, stream>>>(cen, c2, LL * KK);
        dist_wmma_kernel<<<grid, 256, 0, stream>>>(x, cen, x2, c2, out);
    }
}
